// CGD_15685220565300
// MI455X (gfx1250) — matrix operand order repaired and checked by reading, not compiled, not run on hardware
//
#include <hip/hip_runtime.h>
#include <hip/hip_bf16.h>
#include <math.h>

// ---------------------------------------------------------------------------
// Restormer block for MI455X (gfx1250, wave32, WMMA).
// All GEMMs use v_wmma_f32_16x16x32_bf16 (fp32 accumulate).
// Data layout is NCHW throughout; WMMA C-fragments store coalesced NCHW rows.
// Fragment layouts per CDNA5 ISA:
//   A (16x32 bf16): lanes 0-15 -> M=lane, K = {0..7, 16..23};
//                   lanes 16-31 -> M=lane-16, K = {8..15, 24..31}
//   B (32x16 bf16): lanes 0-15 -> N=lane, K = {0..7, 16..23};
//                   lanes 16-31 -> N=lane-16, K = {8..15, 24..31}
//   C/D (16x16 f32): VGPR i, lane l -> M = i + 8*(l>>4), N = l&15
// Workspace arena requirement: ~529 MB.
// ---------------------------------------------------------------------------

typedef __attribute__((ext_vector_type(16))) __bf16       v16bf;
typedef __attribute__((ext_vector_type(8)))  float        v8f;
typedef __attribute__((ext_vector_type(4)))  unsigned int u32x4;

union FragBF { v16bf b; u32x4 u[2]; };

static __device__ __forceinline__ float bf2f(unsigned short h) {
    unsigned u = ((unsigned)h) << 16;
    return __builtin_bit_cast(float, u);
}
static __device__ __forceinline__ unsigned short f2bf(float f) {
    unsigned u = __builtin_bit_cast(unsigned, f);
    u += 0x7FFFu + ((u >> 16) & 1u);   // round-to-nearest-even
    return (unsigned short)(u >> 16);
}

#define NPIX 65536   // 256*256 pixels per batch
#define DIMC 192

// ---------------------------------------------------------------------------
// K0: weight prep (fp32 -> bf16, pad hidden dims) + zero accumulators
// ---------------------------------------------------------------------------
__global__ void prep_kernel(const float* __restrict__ qkv_w,
                            const float* __restrict__ ffn_in_w,
                            const float* __restrict__ ffn_out_w,
                            unsigned short* __restrict__ wqkv,
                            unsigned short* __restrict__ wfi,
                            unsigned short* __restrict__ wfo,
                            float* __restrict__ sumsq,
                            float* __restrict__ G) {
    int tid = blockIdx.x * blockDim.x + threadIdx.x;
    int nt  = gridDim.x * blockDim.x;
    for (int i = tid; i < 576 * 192; i += nt) wqkv[i] = f2bf(qkv_w[i]);
    for (int i = tid; i < 1024 * 192; i += nt) {
        int r = i / 192;
        wfi[i] = (r < 1020) ? f2bf(ffn_in_w[i]) : (unsigned short)0;
    }
    for (int i = tid; i < 192 * 512; i += nt) {
        int r = i >> 9, c = i & 511;
        wfo[i] = (c < 510) ? f2bf(ffn_out_w[r * 510 + c]) : (unsigned short)0;
    }
    for (int i = tid; i < 2 * 384; i += nt) sumsq[i] = 0.f;
    for (int i = tid; i < 2 * 4 * 48 * 48; i += nt) G[i] = 0.f;
}

// ---------------------------------------------------------------------------
// LayerNorm over channels at each pixel (NCHW in, NCHW bf16 out).
// Thread-per-pixel; channel loop gives coalesced loads (consecutive pixels
// in consecutive lanes). Two-pass; source is L2-resident (x fits in 192MB L2).
// ---------------------------------------------------------------------------
__global__ __launch_bounds__(256) void ln_kernel(const float* __restrict__ src,
                                                 const float* __restrict__ w,
                                                 const float* __restrict__ bb,
                                                 unsigned short* __restrict__ dst) {
    int pix = blockIdx.x * 256 + threadIdx.x;
    int b = pix >> 16, p = pix & 65535;
    const float* s = src + (size_t)b * DIMC * NPIX + p;
    float sum = 0.f, sq = 0.f;
    for (int c = 0; c < DIMC; ++c) {
        float v = s[(size_t)c * NPIX];
        sum += v; sq += v * v;
    }
    float mu  = sum * (1.f / DIMC);
    float var = sq * (1.f / DIMC) - mu * mu;
    float rs  = rsqrtf(var + 1e-5f);
    unsigned short* d = dst + (size_t)b * DIMC * NPIX + p;
    for (int c = 0; c < DIMC; ++c) {
        float v = s[(size_t)c * NPIX];
        d[(size_t)c * NPIX] = f2bf((v - mu) * rs * w[c] + bb[c]);
    }
}

// ---------------------------------------------------------------------------
// Generic WMMA GEMM:  Out[b][co][p] = sum_k W[co][k] * Act[b][k][p]
//   W:  bf16 [CO][CK] row-major (optionally per-batch via wstride)
//   Act: bf16 NCHW [B][CK][NPIX]
// Workgroup = 8 waves, macro-tile 64co x 128px, wave tile 32x32 (2x2 WMMA).
// A staged row-major in LDS; B transposed at LDS-write time so fragment
// loads are contiguous 16B/32B ds reads (stride 40 halves keeps alignment).
// Epilogue: C rows are co, cols are consecutive pixels -> coalesced NCHW.
// ---------------------------------------------------------------------------
template <bool STORE_BF16, bool RES>
__global__ __launch_bounds__(256) void gemm_nchw(const unsigned short* __restrict__ Wm,
                                                 const unsigned short* __restrict__ ACT,
                                                 float* __restrict__ outF,
                                                 unsigned short* __restrict__ outH,
                                                 const float* __restrict__ res,
                                                 int CO, int CK, long long wstride) {
    __shared__ unsigned short lA[64 * 40];    // [co][k], stride 40
    __shared__ unsigned short lB[128 * 40];   // [pixel][k], stride 40
    int t = threadIdx.x;
    int tiles_co = CO >> 6;
    int bid = blockIdx.x;
    int pt  = bid & 511;           // 65536/128 = 512 pixel tiles
    int rem = bid >> 9;
    int ct  = rem % tiles_co;
    int b   = rem / tiles_co;
    int co0 = ct * 64, p0 = pt * 128;
    const unsigned short* Wb = Wm + (size_t)b * (size_t)wstride;
    const unsigned short* Ab = ACT + (size_t)b * (size_t)CK * NPIX;

    int lane = t & 31, wid = t >> 5;
    int wco = (wid >> 2) * 32;     // 2 waves along co
    int wp  = (wid & 3) * 32;      // 4 waves along pixels
    int ml = lane & 15, kh = lane >> 4;

    v8f acc[2][2] = {};

    for (int k0 = 0; k0 < CK; k0 += 32) {
        // stage A: 64 rows x 32 k  (1024 u32)
        for (int idx = t; idx < 1024; idx += 256) {
            int r = idx >> 4, j = idx & 15;
            unsigned v = *(const unsigned*)(Wb + (size_t)(co0 + r) * CK + k0 + 2 * j);
            *(unsigned*)(&lA[r * 40 + 2 * j]) = v;
        }
        // stage B transposed: 32 k-rows x 128 px (2048 u32), write [px][k]
        for (int idx = t; idx < 2048; idx += 256) {
            int r = idx >> 6, j = idx & 63;
            unsigned v = *(const unsigned*)(Ab + (size_t)(k0 + r) * NPIX + p0 + 2 * j);
            lB[(2 * j) * 40 + r]     = (unsigned short)v;
            lB[(2 * j + 1) * 40 + r] = (unsigned short)(v >> 16);
        }
        __syncthreads();

        FragBF a[2], bf[2];
        // A fragment: interleaved K halves {kh*8..+8} and {16+kh*8..+8}
#pragma unroll
        for (int mi = 0; mi < 2; ++mi) {
            int row = wco + mi * 16 + ml;
            a[mi].u[0] = *(const u32x4*)&lA[row * 40 + kh * 8];
            a[mi].u[1] = *(const u32x4*)&lA[row * 40 + 16 + kh * 8];
        }
        // B fragment: interleaved K halves {kh*8..+8} and {16+kh*8..+8}
#pragma unroll
        for (int ni = 0; ni < 2; ++ni) {
            int pr = wp + ni * 16 + ml;
            const unsigned short* pb = &lB[pr * 40 + kh * 8];
            bf[ni].u[0] = *(const u32x4*)pb;
            bf[ni].u[1] = *(const u32x4*)(pb + 16);
        }
#pragma unroll
        for (int mi = 0; mi < 2; ++mi)
#pragma unroll
            for (int ni = 0; ni < 2; ++ni)
                acc[mi][ni] = __builtin_amdgcn_wmma_f32_16x16x32_bf16(
                    false, a[mi].b, false, bf[ni].b, (short)0, acc[mi][ni], false, false);
        __syncthreads();
    }

    // epilogue: element (i,lane) -> co = i + 8*(lane>>4), p = lane&15
    size_t obase = (size_t)b * CO * NPIX;
#pragma unroll
    for (int mi = 0; mi < 2; ++mi)
#pragma unroll
        for (int ni = 0; ni < 2; ++ni)
#pragma unroll
            for (int i = 0; i < 8; ++i) {
                int co = co0 + wco + mi * 16 + i + 8 * kh;
                int p  = p0 + wp + ni * 16 + ml;
                size_t o = obase + (size_t)co * NPIX + p;
                float v = acc[mi][ni][i];
                if (RES) v += res[o];
                if (STORE_BF16) outH[o] = f2bf(v);
                else            outF[o] = v;
            }
}

// ---------------------------------------------------------------------------
// Depthwise 3x3 (SAME, zero pad) on qkv (576 ch), bf16 in/out NCHW.
// Splits into q/k/v and accumulates per-channel sum-of-squares for q,k
// (block-reduce + one global f32 atomic per block).
// ---------------------------------------------------------------------------
__global__ __launch_bounds__(256) void dwconv_qkv_kernel(const unsigned short* __restrict__ qkv,
                                                         const float* __restrict__ dww,
                                                         unsigned short* __restrict__ qo,
                                                         unsigned short* __restrict__ ko,
                                                         unsigned short* __restrict__ vo,
                                                         float* __restrict__ sumsq) {
    __shared__ float sred[256];
    int t = threadIdx.x;
    int bid = blockIdx.x;
    int pb = bid & 255;
    int rem = bid >> 8;
    int c = rem % 576, b = rem / 576;
    int p = pb * 256 + t;
    int hh = p >> 8, wp = p & 255;
    const unsigned short* s = qkv + ((size_t)b * 576 + c) * NPIX;
    const float* wv = dww + c * 9;
    float acc = 0.f;
#pragma unroll
    for (int dy = -1; dy <= 1; ++dy) {
        int y = hh + dy;
        if ((unsigned)y < 256u) {
#pragma unroll
            for (int dx = -1; dx <= 1; ++dx) {
                int x = wp + dx;
                if ((unsigned)x < 256u)
                    acc += bf2f(s[y * 256 + x]) * wv[(dy + 1) * 3 + (dx + 1)];
            }
        }
    }
    unsigned short r = f2bf(acc);
    size_t oidx = ((size_t)b * DIMC + (c % DIMC)) * NPIX + p;
    if (c < 192)       qo[oidx] = r;
    else if (c < 384)  ko[oidx] = r;
    else               vo[oidx] = r;

    sred[t] = (c < 384) ? acc * acc : 0.f;
    __syncthreads();
    for (int off = 128; off > 0; off >>= 1) {
        if (t < off) sred[t] += sred[t + off];
        __syncthreads();
    }
    if (t == 0 && c < 384) atomicAdd(&sumsq[b * 384 + c], sred[0]);
}

// ---------------------------------------------------------------------------
// Gram matrices: G[b,h,c,d] = sum_n q[h*48+c, n] * k[h*48+d, n]  (raw, no norm)
// A = q (M=c, K=n), B has column d = k row d: both are spatially contiguous
// in NCHW, so fragments load directly from global.
// K=65536 split into 16 chunks x 4 waves; LDS cross-wave reduce + atomicAdd.
// ---------------------------------------------------------------------------
__global__ __launch_bounds__(128) void gram_kernel(const unsigned short* __restrict__ qb,
                                                   const unsigned short* __restrict__ kb,
                                                   float* __restrict__ G) {
    __shared__ float red[4 * 2304];
    int t = threadIdx.x, lane = t & 31, wid = t >> 5;
    int bid = blockIdx.x;
    int chunk = bid & 15, bh = bid >> 4;     // bh = b*4 + h
    const unsigned short* q = qb + (size_t)bh * 48 * NPIX;
    const unsigned short* k = kb + (size_t)bh * 48 * NPIX;
    int ml = lane & 15, kh = lane >> 4;

    v8f acc[3][3] = {};
    int n0 = chunk * 4096 + wid * 1024;
    for (int it = 0; it < 32; ++it) {
        int n = n0 + it * 32;
        FragBF a[3], bf[3];
        // A fragment: interleaved K halves
#pragma unroll
        for (int mi = 0; mi < 3; ++mi) {
            const unsigned short* r = q + (size_t)(mi * 16 + ml) * NPIX + n + kh * 8;
            a[mi].u[0] = *(const u32x4*)r;
            a[mi].u[1] = *(const u32x4*)(r + 16);
        }
        // B fragment: interleaved K halves
#pragma unroll
        for (int ni = 0; ni < 3; ++ni) {
            const unsigned short* r = k + (size_t)(ni * 16 + ml) * NPIX + n + kh * 8;
            bf[ni].u[0] = *(const u32x4*)r;
            bf[ni].u[1] = *(const u32x4*)(r + 16);
        }
#pragma unroll
        for (int mi = 0; mi < 3; ++mi)
#pragma unroll
            for (int ni = 0; ni < 3; ++ni)
                acc[mi][ni] = __builtin_amdgcn_wmma_f32_16x16x32_bf16(
                    false, a[mi].b, false, bf[ni].b, (short)0, acc[mi][ni], false, false);
    }
#pragma unroll
    for (int mi = 0; mi < 3; ++mi)
#pragma unroll
        for (int ni = 0; ni < 3; ++ni)
#pragma unroll
            for (int i = 0; i < 8; ++i)
                red[wid * 2304 + ((mi * 3 + ni) * 8 + i) * 32 + lane] = acc[mi][ni][i];
    __syncthreads();
    for (int idx = t; idx < 2304; idx += 128) {
        float s = red[idx] + red[2304 + idx] + red[4608 + idx] + red[6912 + idx];
        int l = idx & 31, i = (idx >> 5) & 7, t9 = idx >> 8;
        int mi = t9 / 3, ni = t9 - mi * 3;
        int c = mi * 16 + i + 8 * (l >> 4);
        int d = ni * 16 + (l & 15);
        atomicAdd(&G[bh * 2304 + c * 48 + d], s);
    }
}

// ---------------------------------------------------------------------------
// Softmax(attn) + fold proj:  W_eff[b][o][h*48+d] = sum_c proj[o,h*48+c]*attn
// Tiny (7 MFLOP) -> one block per batch.
// ---------------------------------------------------------------------------
__global__ __launch_bounds__(256) void attn_weff_kernel(const float* __restrict__ G,
                                                        const float* __restrict__ sumsq,
                                                        const float* __restrict__ temp,
                                                        const float* __restrict__ projw,
                                                        unsigned short* __restrict__ weff) {
    __shared__ float at[192 * 48];
    int b = blockIdx.x, t = threadIdx.x;
    if (t < 192) {
        int h = t / 48;
        float nq = fmaxf(sqrtf(sumsq[b * 384 + t]), 1e-12f);
        float tp = temp[h];
        const float* g = G + ((size_t)b * 4 + h) * 2304 + (size_t)(t % 48) * 48;
        float row[48];
        float mx = -3.4e38f;
        for (int d = 0; d < 48; ++d) {
            float nk = fmaxf(sqrtf(sumsq[b * 384 + 192 + h * 48 + d]), 1e-12f);
            float v = g[d] / (nq * nk) * tp;
            row[d] = v;
            mx = fmaxf(mx, v);
        }
        float ssum = 0.f;
        for (int d = 0; d < 48; ++d) { float e = __expf(row[d] - mx); row[d] = e; ssum += e; }
        float inv = 1.f / ssum;
        for (int d = 0; d < 48; ++d) at[t * 48 + d] = row[d] * inv;
    }
    __syncthreads();
    for (int idx = t; idx < 36864; idx += 256) {
        int o = idx / 192, ci = idx % 192;
        int h = ci / 48, d = ci % 48;
        float s = 0.f;
        for (int c2 = 0; c2 < 48; ++c2)
            s += projw[o * 192 + h * 48 + c2] * at[(h * 48 + c2) * 48 + d];
        weff[(size_t)b * 36864 + idx] = f2bf(s);
    }
}

// ---------------------------------------------------------------------------
// FFN depthwise 3x3 + exact GELU(x1)*x2 gate. h has channel stride 1024
// (1020 valid); output g has stride 512 (510 valid, rest zeroed).
// ---------------------------------------------------------------------------
__global__ __launch_bounds__(256) void ffn_dw_gelu_kernel(const unsigned short* __restrict__ hbuf,
                                                          const float* __restrict__ dww,
                                                          unsigned short* __restrict__ g) {
    int t = threadIdx.x;
    int bid = blockIdx.x;
    int pb = bid & 255;
    int rem = bid >> 8;
    int c = rem & 511, b = rem >> 9;
    int p = pb * 256 + t;
    size_t oidx = ((size_t)b * 512 + c) * NPIX + p;
    if (c >= 510) { g[oidx] = 0; return; }
    int hh = p >> 8, wp = p & 255;
    const unsigned short* s1 = hbuf + ((size_t)b * 1024 + c) * NPIX;
    const unsigned short* s2 = hbuf + ((size_t)b * 1024 + 510 + c) * NPIX;
    const float* w1 = dww + c * 9;
    const float* w2 = dww + (510 + c) * 9;
    float a1 = 0.f, a2 = 0.f;
#pragma unroll
    for (int dy = -1; dy <= 1; ++dy) {
        int y = hh + dy;
        if ((unsigned)y < 256u) {
#pragma unroll
            for (int dx = -1; dx <= 1; ++dx) {
                int x = wp + dx;
                if ((unsigned)x < 256u) {
                    int off = y * 256 + x, wi = (dy + 1) * 3 + (dx + 1);
                    a1 += bf2f(s1[off]) * w1[wi];
                    a2 += bf2f(s2[off]) * w2[wi];
                }
            }
        }
    }
    float ge = 0.5f * a1 * (1.f + erff(a1 * 0.70710678118654752f));
    g[oidx] = f2bf(ge * a2);
}

// ---------------------------------------------------------------------------
// Host-side orchestration
// ---------------------------------------------------------------------------
extern "C" void kernel_launch(void* const* d_in, const int* in_sizes, int n_in,
                              void* d_out, int out_size, void* d_ws, size_t ws_size,
                              hipStream_t stream) {
    const float* x        = (const float*)d_in[0];
    const float* ln1_w    = (const float*)d_in[1];
    const float* ln1_b    = (const float*)d_in[2];
    const float* temp     = (const float*)d_in[3];
    const float* qkv_w    = (const float*)d_in[4];
    const float* qkv_dw_w = (const float*)d_in[5];
    const float* proj_w   = (const float*)d_in[6];
    const float* ln2_w    = (const float*)d_in[7];
    const float* ln2_b    = (const float*)d_in[8];
    const float* ffn_in_w = (const float*)d_in[9];
    const float* ffn_dw_w = (const float*)d_in[10];
    const float* ffn_out_w= (const float*)d_in[11];
    float* out = (float*)d_out;

    // Workspace arena (~529 MB), lifetimes allow h to reuse qkv/q/k region.
    char* ws = (char*)d_ws;
    float*          y    = (float*)(ws + 0ULL);                     // [2][192][N] f32
    unsigned short* lnb  = (unsigned short*)(ws + 100663296ULL);    // [2][192][N] bf16 (ln1 then ln2)
    unsigned short* qkvb = (unsigned short*)(ws + 150994944ULL);    // [2][576][N] bf16
    unsigned short* qb   = (unsigned short*)(ws + 301989888ULL);    // [2][192][N] bf16
    unsigned short* kb   = (unsigned short*)(ws + 352321536ULL);    // [2][192][N] bf16
    unsigned short* vb   = (unsigned short*)(ws + 402653184ULL);    // [2][192][N] bf16
    unsigned short* hb   = (unsigned short*)(ws + 150994944ULL);    // [2][1024][N] bf16 (reuse)
    unsigned short* gb   = (unsigned short*)(ws + 419430400ULL);    // [2][512][N] bf16
    unsigned short* wqkv = (unsigned short*)(ws + 553648128ULL);    // [576][192]
    unsigned short* wfi  = wqkv + 576 * 192;                        // [1024][192]
    unsigned short* wfo  = wfi + 1024 * 192;                        // [192][512]
    unsigned short* weff = (unsigned short*)(ws + 554459136ULL);    // [2][192][192]
    float*          ssq  = (float*)(ws + 554606592ULL);             // [2][384]
    float*          G    = (float*)(ws + 554609664ULL);             // [2][4][48][48]

    // 1. weight prep + zero accumulators
    prep_kernel<<<256, 256, 0, stream>>>(qkv_w, ffn_in_w, ffn_out_w, wqkv, wfi, wfo, ssq, G);
    // 2. LN1
    ln_kernel<<<512, 256, 0, stream>>>(x, ln1_w, ln1_b, lnb);
    // 3. qkv = Wqkv @ ln1   (bf16 out)
    gemm_nchw<true, false><<<2 * 9 * 512, 256, 0, stream>>>(wqkv, lnb, nullptr, qkvb, nullptr, 576, 192, 0);
    // 4. depthwise 3x3 -> q,k,v + sum-of-squares for q,k
    dwconv_qkv_kernel<<<2 * 576 * 256, 256, 0, stream>>>(qkvb, qkv_dw_w, qb, kb, vb, ssq);
    // 5. gram matrices (WMMA, K split + atomic reduce)
    gram_kernel<<<8 * 16, 128, 0, stream>>>(qb, kb, G);
    // 6. softmax + fold projection into per-batch W_eff
    attn_weff_kernel<<<2, 256, 0, stream>>>(G, ssq, temp, proj_w, weff);
    // 7. y = W_eff @ v + x   (residual fused)
    gemm_nchw<false, true><<<2 * 3 * 512, 256, 0, stream>>>(weff, vb, y, nullptr, x, 192, 192, 36864);
    // 8. LN2
    ln_kernel<<<512, 256, 0, stream>>>(y, ln2_w, ln2_b, lnb);
    // 9. h = Wfi @ ln2   (1024 padded, bf16 out)
    gemm_nchw<true, false><<<2 * 16 * 512, 256, 0, stream>>>(wfi, lnb, nullptr, hb, nullptr, 1024, 192, 0);
    // 10. depthwise 3x3 + GELU gate -> g (512 padded)
    ffn_dw_gelu_kernel<<<2 * 512 * 256, 256, 0, stream>>>(hb, ffn_dw_w, gb);
    // 11. out = Wfo @ g + y
    gemm_nchw<false, true><<<2 * 3 * 512, 256, 0, stream>>>(wfo, gb, out, nullptr, y, 192, 512, 0);
}
